// MambaSentimentClassifier_19507741459087
// MI455X (gfx1250) — matrix operand order repaired and checked by reading, not compiled, not run on hardware
//
#include <hip/hip_runtime.h>

#define DEV __device__ __forceinline__

typedef __attribute__((ext_vector_type(16))) __bf16 bf16x16;
typedef __attribute__((ext_vector_type(8)))  float  f32x8;

// mamba-130m trimmed config (must match reference)
constexpr int cB  = 2, cL = 1024;
constexpr int DM  = 768;          // d_model
constexpr int DI  = 1536;         // d_inner
constexpr int DS  = 16;           // d_state
constexpr int DTR = 48;           // dt_rank
constexpr int XPN = DTR + 2 * DS; // 80 (x_proj out cols)
constexpr int ROWS = cB * cL;     // 2048
constexpr int DI2  = 2 * DI;      // 3072
constexpr int NUM_LABELS = 2;

// ---------- helpers ----------

DEV unsigned short f2bf(float f) {
  // round-to-nearest-even fp32 -> bf16
  unsigned int u = __float_as_uint(f);
  unsigned int r = u + 0x7FFFu + ((u >> 16) & 1u);
  return (unsigned short)(r >> 16);
}

DEV float block_reduce_sum(float v, float* red) {
  int tid = threadIdx.x;
  red[tid] = v; __syncthreads();
  for (int s = blockDim.x >> 1; s > 0; s >>= 1) {
    if (tid < s) red[tid] += red[tid + s];
    __syncthreads();
  }
  float r = red[0]; __syncthreads();
  return r;
}

DEV bf16x16 pack_frag(uint4 lo, uint4 hi) {
  union { unsigned int u[8]; bf16x16 v; } f;
  f.u[0] = lo.x; f.u[1] = lo.y; f.u[2] = lo.z; f.u[3] = lo.w;
  f.u[4] = hi.x; f.u[5] = hi.y; f.u[6] = hi.z; f.u[7] = hi.w;
  return f.v;
}

// A fragment: 16x32 bf16 (M x K). ISA layout: lanes 0-15 hold rows M=0..15,
// K groups split by lane-half: lane<16 -> K {0..7, 16..23}, lane>=16 -> K {8..15, 24..31}.
// pA points at A + (m0 + lane&15)*K + 8*(lane>>4); k0 added here.
DEV bf16x16 load_frag_A(const unsigned short* __restrict__ pA, int k0) {
  uint4 lo = *(const uint4*)(pA + k0);        // K = k0+8h .. +7
  uint4 hi = *(const uint4*)(pA + k0 + 16);   // K = k0+16+8h .. +7
  return pack_frag(lo, hi);
}

// B fragment: 32x16 bf16 (K x N) taken from row-major weight W[N,K] (B = W^T).
// Lanes hold column N = lane&15; lane<16 -> K {0..7, 16..23}, lane>=16 -> K {8..15, 24..31}.
// pB points at W + (n0 + lane&15)*K + 8*(lane>>4).
DEV bf16x16 load_frag_B(const unsigned short* __restrict__ pB, int k0) {
  uint4 lo = *(const uint4*)(pB + k0);
  uint4 hi = *(const uint4*)(pB + k0 + 16);
  return pack_frag(lo, hi);
}

// ---------- kernels ----------

// C[M,N](fp32) = A[M,K](bf16) * W[N,K]^T(bf16)  (+ Cin if non-null)
// block = 128 threads = 4 waves stacked along M; each wave computes a 16 x (NT*16)
// C tile. NT is compile-time so the inner loop is branch-free (EXEC all-1s) and the
// K-loop is software-pipelined: fragments for slice k+32 load while slice k does WMMA.
template <int NT>
__global__ void __launch_bounds__(128)
gemm_bf16_wmma(const unsigned short* __restrict__ A,
               const unsigned short* __restrict__ W,
               const float* __restrict__ Cin,
               float* __restrict__ Cout,
               int M, int N, int K) {
  const int lane = threadIdx.x & 31;
  const int wave = threadIdx.x >> 5;
  const int m0 = (blockIdx.y * 4 + wave) * 16;
  const int n0 = blockIdx.x * (NT * 16);
  const int h = lane >> 4, r = lane & 15;

  const unsigned short* pA = A + (size_t)(m0 + r) * K + 8 * h;
  const unsigned short* pB[NT];
#pragma unroll
  for (int t = 0; t < NT; ++t)
    pB[t] = W + (size_t)(n0 + t * 16 + r) * K + 8 * h;

  f32x8 acc[NT] = {};

  // pipeline prologue: fragments for k = 0
  bf16x16 a0 = load_frag_A(pA, 0);
  bf16x16 b0[NT];
#pragma unroll
  for (int t = 0; t < NT; ++t) b0[t] = load_frag_B(pB[t], 0);

#pragma unroll 2
  for (int k0 = 32; k0 < K; k0 += 32) {
    // issue next slice's loads before computing on the current one
    bf16x16 a1 = load_frag_A(pA, k0);
    __builtin_prefetch(pA + k0 + 32, 0, 1);
    bf16x16 b1[NT];
#pragma unroll
    for (int t = 0; t < NT; ++t) b1[t] = load_frag_B(pB[t], k0);

#pragma unroll
    for (int t = 0; t < NT; ++t)
      acc[t] = __builtin_amdgcn_wmma_f32_16x16x32_bf16(
          false, a0, false, b0[t], (short)0, acc[t], false, false);

    a0 = a1;
#pragma unroll
    for (int t = 0; t < NT; ++t) b0[t] = b1[t];
  }

  // pipeline epilogue: last slice
#pragma unroll
  for (int t = 0; t < NT; ++t)
    acc[t] = __builtin_amdgcn_wmma_f32_16x16x32_bf16(
        false, a0, false, b0[t], (short)0, acc[t], false, false);

  // C/D layout: VGPR v -> row m0 + v + 8*h, col n0 + lane&15
#pragma unroll
  for (int t = 0; t < NT; ++t) {
    int nt = n0 + t * 16;
#pragma unroll
    for (int v = 0; v < 8; ++v) {
      size_t idx = (size_t)(m0 + v + 8 * h) * N + (nt + r);
      float val = acc[t][v];
      if (Cin) val += Cin[idx];
      Cout[idx] = val;
    }
  }
}

__global__ void embed_gather(const int* __restrict__ ids,
                             const float* __restrict__ emb,
                             float* __restrict__ x) {
  int i = blockIdx.x * 256 + threadIdx.x;
  if (i >= ROWS * DM) return;
  int c = i % DM, row = i / DM;
  x[i] = emb[(size_t)ids[row] * DM + c];
}

__global__ void f32_to_bf16_k(const float* __restrict__ src,
                              unsigned short* __restrict__ dst, int n) {
  int i = blockIdx.x * 256 + threadIdx.x;
  if (i < n) dst[i] = f2bf(src[i]);
}

// one block (256 thr) per row: rmsnorm over DM cols, emit bf16
__global__ void rmsnorm_bf16_k(const float* __restrict__ x,
                               const float* __restrict__ w,
                               unsigned short* __restrict__ out) {
  __shared__ float red[256];
  int row = blockIdx.x, tid = threadIdx.x;
  const float* xr = x + (size_t)row * DM;
  float s = 0.f;
  for (int c = tid; c < DM; c += 256) { float v = xr[c]; s += v * v; }
  s = block_reduce_sum(s, red);
  float scale = rsqrtf(s / DM + 1e-5f);
  for (int c = tid; c < DM; c += 256)
    out[(size_t)row * DM + c] = f2bf(xr[c] * scale * w[c]);
}

// depthwise causal conv (taps=4) + bias + silu; writes u (fp32) and u_bf16
__global__ void conv_silu_k(const float* __restrict__ xz,
                            const float* __restrict__ cw,
                            const float* __restrict__ cb,
                            float* __restrict__ u,
                            unsigned short* __restrict__ ubf) {
  int i = blockIdx.x * 256 + threadIdx.x;
  if (i >= ROWS * DI) return;
  int d = i % DI, row = i / DI;
  int l = row % cL, b = row / cL;
  float s = cb[d];
#pragma unroll
  for (int j = 0; j < 4; ++j) {
    int ll = l - 3 + j;
    if (ll >= 0) s += cw[d * 4 + j] * xz[(size_t)(b * cL + ll) * DI2 + d];
  }
  float r = s / (1.f + __expf(-s));  // silu
  u[i] = r;
  ubf[i] = f2bf(r);
}

// dt = softplus(dt_lowrank @ dt_proj_w^T + b), dt row (48) staged in LDS
__global__ void dt_softplus_k(const float* __restrict__ xdbl,
                              const float* __restrict__ wdt,
                              const float* __restrict__ bdt,
                              float* __restrict__ dtf) {
  __shared__ float dts[DTR];
  int row = blockIdx.x, tid = threadIdx.x;
  if (tid < DTR) dts[tid] = xdbl[(size_t)row * XPN + tid];
  __syncthreads();
  for (int d = tid; d < DI; d += 256) {
    float s = bdt[d];
    const float* wr = wdt + (size_t)d * DTR;
#pragma unroll
    for (int r = 0; r < DTR; ++r) s += dts[r] * wr[r];
    dtf[(size_t)row * DI + d] = (s > 20.f) ? s : log1pf(__expf(s));
  }
}

__global__ void aneg_k(const float* __restrict__ alog, float* __restrict__ an, int n) {
  int i = blockIdx.x * 256 + threadIdx.x;
  if (i < n) an[i] = -__expf(alog[i]);
}

// selective scan: 1 thread per (b,d), 16 states in registers, B/C rows in LDS.
// Fused epilogue: y = sum(h*C) + u*D; yy = y * silu(z); write bf16 for out_proj.
__global__ void __launch_bounds__(256)
scan_k(const float* __restrict__ dtf, const float* __restrict__ u,
       const float* __restrict__ xdbl, const float* __restrict__ xz,
       const float* __restrict__ Aneg, const float* __restrict__ Dv,
       unsigned short* __restrict__ yybf) {
  __shared__ float BC[2 * DS];             // B row then C row
  int b = blockIdx.x / (DI / 256);
  int d = (blockIdx.x % (DI / 256)) * 256 + threadIdx.x;

  float h[DS], Ar[DS];
#pragma unroll
  for (int n = 0; n < DS; ++n) { h[n] = 0.f; Ar[n] = Aneg[(size_t)d * DS + n]; }
  float dd = Dv[d];

  for (int t = 0; t < cL; ++t) {
    size_t row = (size_t)b * cL + t;
    if (threadIdx.x < 2 * DS)
      BC[threadIdx.x] = xdbl[row * XPN + DTR + threadIdx.x];
    __syncthreads();
    float dt = dtf[row * DI + d];
    float uu = u[row * DI + d];
    float du = dt * uu;
    float y = 0.f;
#pragma unroll
    for (int n = 0; n < DS; ++n) {
      float dA = __expf(dt * Ar[n]);
      h[n] = dA * h[n] + du * BC[n];
      y += h[n] * BC[DS + n];
    }
    y += uu * dd;
    float zz = xz[row * DI2 + DI + d];
    float yy = y * (zz / (1.f + __expf(-zz)));
    yybf[row * DI + d] = f2bf(yy);
    __syncthreads();
  }
}

// final rmsnorm (last token) + classifier head; one block per batch element
__global__ void head_k(const float* __restrict__ x, const float* __restrict__ nf,
                       const float* __restrict__ cw, const float* __restrict__ cb,
                       float* __restrict__ out) {
  __shared__ float red[256];
  int b = blockIdx.x, tid = threadIdx.x;
  const float* xr = x + (size_t)(b * cL + (cL - 1)) * DM;
  float s = 0.f;
  for (int c = tid; c < DM; c += 256) { float v = xr[c]; s += v * v; }
  s = block_reduce_sum(s, red);
  float scale = rsqrtf(s / DM + 1e-5f);
  for (int j = 0; j < NUM_LABELS; ++j) {
    float p = 0.f;
    for (int c = tid; c < DM; c += 256)
      p += xr[c] * scale * nf[c] * cw[(size_t)j * DM + c];
    p = block_reduce_sum(p, red);
    if (tid == 0) out[b * NUM_LABELS + j] = p + cb[j];
  }
}

// ---------- launch ----------

extern "C" void kernel_launch(void* const* d_in, const int* in_sizes, int n_in,
                              void* d_out, int out_size, void* d_ws, size_t ws_size,
                              hipStream_t stream) {
  const int*   ids      = (const int*)  d_in[0];
  const float* emb      = (const float*)d_in[1];
  const float* norm_w   = (const float*)d_in[2];   // [4, DM]
  const float* in_proj  = (const float*)d_in[3];   // [4, DI2, DM]
  const float* conv_w   = (const float*)d_in[4];   // [4, DI, 1, 4]
  const float* conv_b   = (const float*)d_in[5];   // [4, DI]
  const float* x_proj   = (const float*)d_in[6];   // [4, XPN, DI]
  const float* dt_w     = (const float*)d_in[7];   // [4, DI, DTR]
  const float* dt_b     = (const float*)d_in[8];   // [4, DI]
  const float* A_log    = (const float*)d_in[9];   // [4, DI, DS]
  const float* Dp       = (const float*)d_in[10];  // [4, DI]
  const float* out_proj = (const float*)d_in[11];  // [4, DM, DI]
  const float* norm_f   = (const float*)d_in[12];  // [DM]
  const float* cls_w    = (const float*)d_in[13];  // [2, DM]
  const float* cls_b    = (const float*)d_in[14];  // [2]
  float* out = (float*)d_out;

  char* ws = (char*)d_ws;
  size_t off = 0;
  auto alloc = [&](size_t bytes) -> char* {
    char* p = ws + off;
    off = (off + bytes + 255) & ~(size_t)255;
    return p;
  };

  float*          x      = (float*)alloc((size_t)ROWS * DM  * 4);
  float*          xz     = (float*)alloc((size_t)ROWS * DI2 * 4);
  float*          u      = (float*)alloc((size_t)ROWS * DI  * 4);
  float*          dtf    = (float*)alloc((size_t)ROWS * DI  * 4);
  float*          xdbl   = (float*)alloc((size_t)ROWS * XPN * 4);
  float*          Aneg   = (float*)alloc((size_t)DI * DS * 4);
  unsigned short* xn_bf  = (unsigned short*)alloc((size_t)ROWS * DM  * 2);
  unsigned short* u_bf   = (unsigned short*)alloc((size_t)ROWS * DI  * 2);
  unsigned short* yy_bf  = (unsigned short*)alloc((size_t)ROWS * DI  * 2);
  unsigned short* win_bf = (unsigned short*)alloc((size_t)DI2 * DM * 2);
  unsigned short* wxp_bf = (unsigned short*)alloc((size_t)XPN * DI * 2);
  unsigned short* wop_bf = (unsigned short*)alloc((size_t)DM  * DI * 2);

  auto grid1 = [](int n) { return dim3((n + 255) / 256); };

  // embedding gather
  embed_gather<<<grid1(ROWS * DM), 256, 0, stream>>>(ids, emb, x);

  for (int i = 0; i < 4; ++i) {
    // rmsnorm -> bf16 activations
    rmsnorm_bf16_k<<<ROWS, 256, 0, stream>>>(x, norm_w + (size_t)i * DM, xn_bf);

    // in_proj GEMM (WMMA): xz[2048,3072] = xn[2048,768] @ Win^T
    f32_to_bf16_k<<<grid1(DI2 * DM), 256, 0, stream>>>(
        in_proj + (size_t)i * DI2 * DM, win_bf, DI2 * DM);
    gemm_bf16_wmma<4><<<dim3(DI2 / 64, ROWS / 64), 128, 0, stream>>>(
        xn_bf, win_bf, nullptr, xz, ROWS, DI2, DM);

    // depthwise causal conv + silu
    conv_silu_k<<<grid1(ROWS * DI), 256, 0, stream>>>(
        xz, conv_w + (size_t)i * DI * 4, conv_b + (size_t)i * DI, u, u_bf);

    // x_proj GEMM (WMMA): xdbl[2048,80] = u[2048,1536] @ Wxp^T  (80 = 5 n-tiles)
    f32_to_bf16_k<<<grid1(XPN * DI), 256, 0, stream>>>(
        x_proj + (size_t)i * XPN * DI, wxp_bf, XPN * DI);
    gemm_bf16_wmma<5><<<dim3(1, ROWS / 64), 128, 0, stream>>>(
        u_bf, wxp_bf, nullptr, xdbl, ROWS, XPN, DI);

    // dt projection + softplus (K=48 too skinny for WMMA -> VALU with LDS row)
    dt_softplus_k<<<ROWS, 256, 0, stream>>>(
        xdbl, dt_w + (size_t)i * DI * DTR, dt_b + (size_t)i * DI, dtf);

    // A = -exp(A_log)
    aneg_k<<<grid1(DI * DS), 256, 0, stream>>>(
        A_log + (size_t)i * DI * DS, Aneg, DI * DS);

    // selective scan (fused D-residual, silu(z) gate, bf16 emit)
    scan_k<<<cB * (DI / 256), 256, 0, stream>>>(
        dtf, u, xdbl, xz, Aneg, Dp + (size_t)i * DI, yy_bf);

    // out_proj GEMM (WMMA) with residual: x += yy @ Wout^T
    f32_to_bf16_k<<<grid1(DM * DI), 256, 0, stream>>>(
        out_proj + (size_t)i * DM * DI, wop_bf, DM * DI);
    gemm_bf16_wmma<4><<<dim3(DM / 64, ROWS / 64), 128, 0, stream>>>(
        yy_bf, wop_bf, x, x, ROWS, DM, DI);
  }

  // final norm + classifier head (last token)
  head_k<<<cB, 256, 0, stream>>>(x, norm_f, cls_w, cls_b, out);
}
